// DPR_48584670053100
// MI455X (gfx1250) — matrix operand order repaired and checked by reading, not compiled, not run on hardware
//
#include <hip/hip_runtime.h>
#include <math.h>

typedef __attribute__((ext_vector_type(16))) _Float16 v16h;
typedef __attribute__((ext_vector_type(8)))  float    v8f;

#define NB    128            // B (queries)
#define NP    256            // P (passages)
#define MQ    16             // query tokens
#define MP    32             // passage tokens
#define DDIM  768            // embedding dim
#define KSTEP 32             // WMMA K per step (f16)
#define NKS   (DDIM / KSTEP) // 24
#define PT    8              // passages per workgroup (1 per wave)
#define DPAD  (DDIM + 8)     // padded LDS row stride in halfs (bank-conflict-free)

// ---------- float <-> sortable-uint (monotonic total order) ----------
__device__ __forceinline__ unsigned f2s(float f) {
  unsigned u = __float_as_uint(f);
  return u ^ (unsigned)(((int)u >> 31) | 0x80000000);
}
__device__ __forceinline__ float s2f(unsigned s) {
  unsigned u = (s & 0x80000000u) ? (s ^ 0x80000000u) : ~s;
  return __uint_as_float(u);
}

// ---------- wave32 reductions ----------
__device__ __forceinline__ int wred_i(int x) {
#pragma unroll
  for (int off = 16; off >= 1; off >>= 1) x += __shfl_xor(x, off, 32);
  return x;
}
__device__ __forceinline__ float wred_f(float x) {
#pragma unroll
  for (int off = 16; off >= 1; off >>= 1) x += __shfl_xor(x, off, 32);
  return x;
}

// Dual exact selection over the wave's 512-element tile (16 values/lane):
//   top  = sum of k1 largest valid values
//   bot  = sum of (n - k2) smallest valid values = tot - sum of k2 largest
// Single bit-radix binary search finds both thresholds at once; the two
// wave-wide counts are packed into one register (each <= 512, no carry).
// Exact tie handling: sum(v >= T) - (cnt - k) * T.
__device__ __forceinline__ void dual_topk(const float* vals, unsigned vmask,
                                          int k1, int k2, float tot,
                                          float& top, float& bot) {
  unsigned su[16];
#pragma unroll
  for (int i = 0; i < 16; ++i)
    su[i] = ((vmask >> i) & 1u) ? f2s(vals[i]) : 0u;  // invalid -> below any valid key

  unsigned T1 = 0u, T2 = 0u;
  for (int bit = 31; bit >= 0; --bit) {
    const unsigned m  = 1u << bit;
    const unsigned t1 = T1 | m, t2 = T2 | m;
    int c = 0;
#pragma unroll
    for (int i = 0; i < 16; ++i) {
      c += (su[i] >= t1) ? 1 : 0;
      c += (su[i] >= t2) ? (1 << 16) : 0;
    }
    c = wred_i(c);                       // uniform across the wave
    if ((c & 0xFFFF) >= k1) T1 = t1;
    if ((c >> 16)    >= k2) T2 = t2;
  }

  int cc = 0;
  float s1 = 0.f, s2 = 0.f;
#pragma unroll
  for (int i = 0; i < 16; ++i) {
    if (su[i] >= T1) { cc += 1;         s1 += vals[i]; }
    if (su[i] >= T2) { cc += (1 << 16); s2 += vals[i]; }
  }
  cc = wred_i(cc);
  s1 = wred_f(s1);
  s2 = wred_f(s2);

  const int c1 = cc & 0xFFFF, c2 = cc >> 16;
  top = s1 - (float)(c1 - k1) * s2f(T1);                            // k1 >= 1 always
  const float topk2 = (k2 > 0) ? (s2 - (float)(c2 - k2) * s2f(T2)) : 0.f;
  bot = tot - topk2;
}

// ---------- fragment loaders ----------
// B fragment (K=32 x N=16, f16): lane L -> column n=L&15, K range [16*(L>>4), +16)
// contiguous.  Pre-converted path: 32B contiguous per lane.
__device__ __forceinline__ v16h load_frag_h(const _Float16* base) {
  union { uint4 u[2]; v16h h; } c;
  c.u[0] = ((const uint4*)base)[0];
  c.u[1] = ((const uint4*)base)[1];
  return c.h;
}
__device__ __forceinline__ v16h load_frag_f32(const float* base) {
  float t[16];
#pragma unroll
  for (int i = 0; i < 4; ++i) {
    float4 f = ((const float4*)base)[i];
    t[4 * i + 0] = f.x; t[4 * i + 1] = f.y; t[4 * i + 2] = f.z; t[4 * i + 3] = f.w;
  }
  v16h r;
#pragma unroll
  for (int i = 0; i < 16; ++i) r[i] = (_Float16)t[i];
  return r;
}

// ---------- prep: convert p_emb f32 -> f16 and relayout into B-fragment order
// ws layout: chunk c = ((p*NKS + kk)*2 + tile)*32 + lane, 16 halfs (32B) per chunk,
// so the GEMM hot loop issues fully coalesced b128 loads.
__global__ __launch_bounds__(256) void dpr_prep_kernel(const float* __restrict__ p_emb,
                                                       void* __restrict__ ws) {
  _Float16* ph = (_Float16*)ws;
  const int c    = blockIdx.x * 256 + threadIdx.x;  // exact multiple, no bounds check
  const int lane = c & 31;
  const int tile = (c >> 5) & 1;
  const int rest = c >> 6;         // p*NKS + kk
  const int kk   = rest % NKS;
  const int p    = rest / NKS;
  const float* src = p_emb +
      (size_t)(p * MP + (lane & 15) + 16 * tile) * DDIM + kk * KSTEP + 16 * (lane >> 4);
  union { uint4 u[2]; _Float16 h[16]; } w;
#pragma unroll
  for (int i = 0; i < 16; ++i) w.h[i] = (_Float16)src[i];
  uint4* dst = (uint4*)(ph + (size_t)c * 16);
  dst[0] = w.u[0];
  dst[1] = w.u[1];
}

// ---------- main fused kernel: WMMA GEMM + masked dual top-k -> logits ----------
template <bool PRE>
__global__ __launch_bounds__(256) void dpr_main_kernel(
    const float* __restrict__ q_emb, const void* __restrict__ p_src,
    const unsigned char* __restrict__ q_mask, const unsigned char* __restrict__ p_mask,
    const float* __restrict__ alpha_raw, const float* __restrict__ beta_raw,
    float* __restrict__ logits) {
  __shared__ __align__(16) _Float16 qsh[MQ * DPAD];

  const int tid  = threadIdx.x;
  const int b    = blockIdx.y;
  const int wave = tid >> 5;
  const int lane = tid & 31;
  const int p    = blockIdx.x * PT + wave;

  // load + convert this block's query (16 x 768) into LDS once
  const float* qbase = q_emb + (size_t)b * MQ * DDIM;
  for (int r = 0; r < MQ; ++r)
    for (int c = tid; c < DDIM; c += 256)
      qsh[r * DPAD + c] = (_Float16)qbase[r * DDIM + c];
  __syncthreads();

  const int mrow = lane & 15;   // A row / B column / C column index
  const int hi   = lane >> 4;   // lane half selector

  // A fragment (16x32 f16): lane holds K in {koff..koff+15}, koff=16*hi
  const _Float16* aRow = qsh + mrow * DPAD + 16 * hi;

  v8f c0 = {}, c1 = {};

  if (PRE) {
    const _Float16* ph = (const _Float16*)p_src;
    const _Float16* pb = ph + (size_t)p * (NKS * 2 * 512) + lane * 16;
    for (int kk = 0; kk < NKS; ++kk) {
      union { uint4 u[2]; v16h h; } ua;
      ua.u[0] = ((const uint4*)(aRow + kk * KSTEP))[0];
      ua.u[1] = ((const uint4*)(aRow + kk * KSTEP + 8))[0];
      v16h a  = ua.h;
      v16h b0 = load_frag_h(pb + (kk * 2 + 0) * 512);
      v16h b1 = load_frag_h(pb + (kk * 2 + 1) * 512);
      c0 = __builtin_amdgcn_wmma_f32_16x16x32_f16(false, a, false, b0, (short)0, c0, false, false);
      c1 = __builtin_amdgcn_wmma_f32_16x16x32_f16(false, a, false, b1, (short)0, c1, false, false);
    }
  } else {
    const float* pe    = (const float*)p_src;
    const float* b0Row = pe + (size_t)(p * MP + mrow) * DDIM + 16 * hi;
    const float* b1Row = pe + (size_t)(p * MP + mrow + 16) * DDIM + 16 * hi;
    for (int kk = 0; kk < NKS; ++kk) {
      union { uint4 u[2]; v16h h; } ua;
      ua.u[0] = ((const uint4*)(aRow + kk * KSTEP))[0];
      ua.u[1] = ((const uint4*)(aRow + kk * KSTEP + 8))[0];
      v16h a  = ua.h;
      v16h b0 = load_frag_f32(b0Row + kk * KSTEP);
      v16h b1 = load_frag_f32(b1Row + kk * KSTEP);
      c0 = __builtin_amdgcn_wmma_f32_16x16x32_f16(false, a, false, b0, (short)0, c0, false, false);
      c1 = __builtin_amdgcn_wmma_f32_16x16x32_f16(false, a, false, b1, (short)0, c1, false, false);
    }
  }

  // validity masks via wave32 ballot (index with lane&15 to stay in bounds)
  const unsigned qbits =
      (unsigned)__ballot((lane < MQ) && (q_mask[b * MQ + (lane & 15)] != 0)) & 0xFFFFu;
  const unsigned pbits = (unsigned)__ballot(p_mask[p * MP + lane] != 0);

  const int n  = __popc(qbits) * __popc(pbits);
  const int k1 = max((4 * n) / 10, 1);          // k  (top count)
  const int l  = min((8 * n) / 10, n);          // l  (bottom count)
  const int k2 = n - l;                         // bottom via tot - top_{n-l}

  // C layout: lane holds, for r=0..7: sim[M=r+8*hi][N=mrow] (c0) and [N=mrow+16] (c1)
  float vals[16];
  unsigned vmask = 0;
#pragma unroll
  for (int r = 0; r < 8; ++r) {
    vals[r]     = c0[r];
    vals[8 + r] = c1[r];
    const unsigned qv = (qbits >> (r + 8 * hi)) & 1u;
    const unsigned p0 = (pbits >> mrow) & 1u;
    const unsigned p1 = (pbits >> (mrow + 16)) & 1u;
    vmask |= (qv & p0) << r;
    vmask |= (qv & p1) << (8 + r);
  }

  float tot = 0.f;
#pragma unroll
  for (int i = 0; i < 16; ++i) tot += ((vmask >> i) & 1u) ? vals[i] : 0.f;
  tot = wred_f(tot);

  float top_sum, bottom_sum;
  dual_topk(vals, vmask, k1, k2, tot, top_sum, bottom_sum);

  if (lane == 0) {
    const float ar = alpha_raw[0], br = beta_raw[0];
    const float alpha = (ar > 20.f) ? ar : log1pf(expf(ar));
    const float beta  = (br > 20.f) ? br : log1pf(expf(br));
    logits[b * NP + p] = alpha * top_sum - beta * bottom_sum;
  }
}

// ---------- cross-entropy with labels = arange(B) ----------
__global__ __launch_bounds__(256) void dpr_loss_kernel(const float* __restrict__ logits,
                                                       float* __restrict__ out) {
  __shared__ float red[256];
  const int t = threadIdx.x;
  float acc = 0.f;
  for (int b = 0; b < NB; ++b) {
    const float* row = logits + b * NP;   // NP == blockDim == 256
    red[t] = row[t];
    __syncthreads();
#pragma unroll
    for (int off = 128; off >= 1; off >>= 1) {
      if (t < off) red[t] = fmaxf(red[t], red[t + off]);
      __syncthreads();
    }
    const float m = red[0];
    __syncthreads();
    red[t] = expf(row[t] - m);
    __syncthreads();
#pragma unroll
    for (int off = 128; off >= 1; off >>= 1) {
      if (t < off) red[t] += red[t + off];
      __syncthreads();
    }
    if (t == 0) acc += m + logf(red[0]) - row[b];
    __syncthreads();
  }
  if (t == 0) out[0] = acc / (float)NB;
}

extern "C" void kernel_launch(void* const* d_in, const int* in_sizes, int n_in,
                              void* d_out, int out_size, void* d_ws, size_t ws_size,
                              hipStream_t stream) {
  (void)in_sizes; (void)n_in; (void)out_size;
  const float* q_emb         = (const float*)d_in[0];
  const float* p_emb         = (const float*)d_in[1];
  const unsigned char* qmask = (const unsigned char*)d_in[2];
  const unsigned char* pmask = (const unsigned char*)d_in[3];
  const float* alpha_raw     = (const float*)d_in[4];
  const float* beta_raw      = (const float*)d_in[5];

  float* out    = (float*)d_out;
  float* logits = out + 1;   // d_out = [loss, logits(B*P)] in return order

  const size_t ph_bytes = (size_t)NP * MP * DDIM * sizeof(unsigned short);  // ~12.6 MB
  dim3 grid(NP / PT, NB);

  if (ws_size >= ph_bytes) {
    const int chunks = NP * NKS * 2 * 32;  // 393216, multiple of 256
    dpr_prep_kernel<<<chunks / 256, 256, 0, stream>>>(p_emb, d_ws);
    dpr_main_kernel<true><<<grid, 256, 0, stream>>>(q_emb, d_ws, qmask, pmask,
                                                    alpha_raw, beta_raw, logits);
  } else {
    dpr_main_kernel<false><<<grid, 256, 0, stream>>>(q_emb, p_emb, qmask, pmask,
                                                     alpha_raw, beta_raw, logits);
  }
  dpr_loss_kernel<<<1, 256, 0, stream>>>(logits, out);
}
